// FastAttention_53343493816385
// MI455X (gfx1250) — matrix operand order repaired and checked by reading, not compiled, not run on hardware
//
#include <hip/hip_runtime.h>

// ---------------------------------------------------------------------------
// LinearAttention pipeline for MI455X (gfx1250, wave32, WMMA f16 -> f32 acc)
//   B=16, C=256, HEADS=8, DIM_HEAD=64, HIDDEN=512, n=H*W=4096
// Workspace layout (halves):
//   qkv_f16 : [16][1536][4096]   (q rows 0..511 pre-scaled, k 512..1023, v 1024..1535)
//   ctxT_f16: [16][8][64][64]    (context transposed: ctxT[e][d])
//   attention output reuses the v region of qkv_f16 (v is dead after context GEMM)
// ---------------------------------------------------------------------------

typedef __attribute__((ext_vector_type(16))) _Float16 v16h;
typedef __attribute__((ext_vector_type(8)))  _Float16 h8;
typedef __attribute__((ext_vector_type(8)))  float    v8f;
typedef __attribute__((ext_vector_type(4)))  float    f32x4;

#define NB   16
#define NCH  256
#define NH   8
#define DH   64
#define HID  512
#define NSEQ 4096
#define QKV_M 1536
#define ATT_SCALE 0.125f   // 64^-0.5

__device__ __forceinline__ v16h frag16(h8 lo, h8 hi) {
  return __builtin_shufflevector(lo, hi, 0,1,2,3,4,5,6,7,8,9,10,11,12,13,14,15);
}
__device__ __forceinline__ v8f wmma16(v16h a, v16h b, v8f c) {
  // v_wmma_f32_16x16x32_f16: (neg_a, A, neg_b, B, c_mod, C, reuse_a, reuse_b)
  return __builtin_amdgcn_wmma_f32_16x16x32_f16(false, a, false, b, (short)0, c,
                                                false, false);
}

// ---------------------------------------------------------------------------
// Generic tiled WMMA GEMM:  C[b] = A[M,K] (fp32) x B[b][K,N]  (+epilogue)
// Tile 128x128, BK=32, 256 threads = 8 waves, each wave 32x64 (2x4 subtiles).
// Register-prefetch pipeline: next K-tile's global loads are issued before the
// WMMAs on the current LDS tile so HBM latency overlaps matrix math.
// ---------------------------------------------------------------------------
template<bool B_IS_F16, bool EPI_QKV>
__global__ __launch_bounds__(256)
void gemm_wmma_kernel(const float* __restrict__ A, const void* __restrict__ Bv,
                      void* __restrict__ Cv, const float* __restrict__ bias,
                      int M, int N, int K, long strideB, long strideC) {
  __shared__ __align__(16) _Float16 As[128][40];  // [m][k], 80B row stride
  __shared__ __align__(16) _Float16 Bs[128][40];  // [n][k] (transposed tile)

  const int tid  = threadIdx.x;
  const int wave = tid >> 5, lane = tid & 31;
  const int wm = wave & 3, wn = wave >> 2;
  const int lrow = lane & 15;
  const int ah = (lane < 16) ? 0 : 8;    // A-fragment K base (ISA 7.12.2)
  const int bh = (lane < 16) ? 0 : 16;   // unused; B fragment is read with ah, same K order as A
  const long bm = (long)blockIdx.y * 128;
  const long bn = (long)blockIdx.x * 128;

  const float*    Bf = (const float*)Bv    + (long)blockIdx.z * strideB;
  const _Float16* Bh = (const _Float16*)Bv + (long)blockIdx.z * strideB;

  // register staging for the global->LDS pipeline
  f32x4 aReg[4];
  f32x4 bRegF[4];
  h8    bRegH[2];

  auto loadA = [&](int k0) {
#pragma unroll
    for (int i = 0; i < 4; ++i) {
      int f = tid + i * 256;
      int r = f >> 3, c4 = (f & 7) << 2;
      aReg[i] = *(const f32x4*)(A + (bm + r) * (long)K + k0 + c4);
    }
  };
  auto loadB = [&](int k0) {
    if constexpr (!B_IS_F16) {
#pragma unroll
      for (int i = 0; i < 4; ++i) {
        int f = tid + i * 256;
        int k = f >> 5, n4 = (f & 31) << 2;
        bRegF[i] = *(const f32x4*)(Bf + (long)(k0 + k) * N + bn + n4);
      }
    } else {
#pragma unroll
      for (int i = 0; i < 2; ++i) {
        int f = tid + i * 256;
        int k = f >> 4, n8 = (f & 15) << 3;
        bRegH[i] = *(const h8*)(Bh + (long)(k0 + k) * N + bn + n8);
      }
    }
  };
  auto stage = [&]() {
#pragma unroll
    for (int i = 0; i < 4; ++i) {
      int f = tid + i * 256;
      int r = f >> 3, c4 = (f & 7) << 2;
      As[r][c4 + 0] = (_Float16)aReg[i][0];
      As[r][c4 + 1] = (_Float16)aReg[i][1];
      As[r][c4 + 2] = (_Float16)aReg[i][2];
      As[r][c4 + 3] = (_Float16)aReg[i][3];
    }
    if constexpr (!B_IS_F16) {
#pragma unroll
      for (int i = 0; i < 4; ++i) {
        int f = tid + i * 256;
        int k = f >> 5, n4 = (f & 31) << 2;
        Bs[n4 + 0][k] = (_Float16)bRegF[i][0];
        Bs[n4 + 1][k] = (_Float16)bRegF[i][1];
        Bs[n4 + 2][k] = (_Float16)bRegF[i][2];
        Bs[n4 + 3][k] = (_Float16)bRegF[i][3];
      }
    } else {
#pragma unroll
      for (int i = 0; i < 2; ++i) {
        int f = tid + i * 256;
        int k = f >> 4, n8 = (f & 15) << 3;
#pragma unroll
        for (int j = 0; j < 8; ++j) Bs[n8 + j][k] = bRegH[i][j];
      }
    }
  };

  v8f acc[2][4] = {};

  loadA(0);
  loadB(0);
  for (int k0 = 0; k0 < K; k0 += 32) {
    stage();                       // waits current loads, fills LDS
    __syncthreads();
    if (k0 + 32 < K) {             // next tile's loads fly during the WMMAs
      loadA(k0 + 32);
      loadB(k0 + 32);
      if constexpr (!B_IS_F16)
        __builtin_prefetch(Bf + (long)(k0 + 64) * N + bn, 0, 3);
      else
        __builtin_prefetch(Bh + (long)(k0 + 64) * N + bn, 0, 3);
    }

    v16h af[2], bf[4];
#pragma unroll
    for (int s = 0; s < 2; ++s) {
      int m = wm * 32 + s * 16 + lrow;
      af[s] = frag16(*(const h8*)&As[m][ah], *(const h8*)&As[m][ah + 16]);
    }
#pragma unroll
    for (int t = 0; t < 4; ++t) {
      int n = wn * 64 + t * 16 + lrow;
      bf[t] = frag16(*(const h8*)&Bs[n][ah], *(const h8*)&Bs[n][ah + 16]);
    }
#pragma unroll
    for (int s = 0; s < 2; ++s)
#pragma unroll
      for (int t = 0; t < 4; ++t)
        acc[s][t] = wmma16(af[s], bf[t], acc[s][t]);
    __syncthreads();
  }

  // ---- epilogue: C lane layout: VGPR r -> M = r + (lane>=16 ? 8:0), N = lane&15
#pragma unroll
  for (int s = 0; s < 2; ++s) {
#pragma unroll
    for (int t = 0; t < 4; ++t) {
#pragma unroll
      for (int r = 0; r < 8; ++r) {
        long gm = bm + wm * 32 + s * 16 + r + ah;
        long gn = bn + wn * 64 + t * 16 + lrow;
        float val = acc[s][t][r];
        if constexpr (EPI_QKV) {
          _Float16* Cp = (_Float16*)Cv + (long)blockIdx.z * strideC;
          if (gm < HID) val *= ATT_SCALE;   // q rows pre-scaled
          Cp[gm * (long)N + gn] = (_Float16)val;
        } else {
          float* Cp = (float*)Cv + (long)blockIdx.z * strideC;
          Cp[gm * (long)N + gn] = val + bias[gm];
        }
      }
    }
  }
}

// ---------------------------------------------------------------------------
// Softmax over n=4096, in place on the k region (f16 storage, f32 math).
// One block (256 threads) per row; 16 elements per thread.
// ---------------------------------------------------------------------------
__global__ __launch_bounds__(256)
void softmax_rows_kernel(_Float16* __restrict__ qkv) {
  const int tid = threadIdx.x;
  const long row = blockIdx.x;           // b*512 + r
  const long b = row >> 9;
  const int  r = (int)(row & 511);
  _Float16* p = qkv + ((b * QKV_M) + HID + r) * (long)NSEQ;

  float vals[16];
  float m = -1e30f;
#pragma unroll
  for (int i = 0; i < 16; ++i) {
    vals[i] = (float)p[tid + i * 256];
    m = fmaxf(m, vals[i]);
  }
  __shared__ float red[256];
  red[tid] = m;
  __syncthreads();
  for (int s = 128; s > 0; s >>= 1) {
    if (tid < s) red[tid] = fmaxf(red[tid], red[tid + s]);
    __syncthreads();
  }
  const float rowmax = red[0];
  __syncthreads();
  float sum = 0.f;
#pragma unroll
  for (int i = 0; i < 16; ++i) {
    vals[i] = __expf(vals[i] - rowmax);
    sum += vals[i];
  }
  red[tid] = sum;
  __syncthreads();
  for (int s = 128; s > 0; s >>= 1) {
    if (tid < s) red[tid] += red[tid + s];
    __syncthreads();
  }
  const float inv = 1.f / red[0];
#pragma unroll
  for (int i = 0; i < 16; ++i)
    p[tid + i * 256] = (_Float16)(vals[i] * inv);
}

// ---------------------------------------------------------------------------
// Context: ctx[d,e] = sum_n k[d,n] * v[e,n], per (b,h). Both operands are
// row-major over n == exactly the WMMA fragment layouts -> direct global 16B
// fragment loads. 2x-unrolled ping-pong register pipeline: loads for one
// buffer fly while WMMAs consume the other; no register copies, no WAR NOPs.
// 128 threads = 4 waves; wave w owns d-strip w*16.., 4 e-subtiles each.
// ---------------------------------------------------------------------------
__global__ __launch_bounds__(128)
void context_kernel(const _Float16* __restrict__ qkv, _Float16* __restrict__ ctxT) {
  const int h = blockIdx.x, b = blockIdx.y;
  const _Float16* kp = qkv + ((long)b * QKV_M + HID + h * DH) * NSEQ;
  const _Float16* vp = qkv + ((long)b * QKV_M + 2 * HID + h * DH) * NSEQ;
  const int wave = threadIdx.x >> 5, lane = threadIdx.x & 31;
  const int lrow = lane & 15;
  const int ah = (lane < 16) ? 0 : 8;
  const int bh = (lane < 16) ? 0 : 16;

  const _Float16* arow  = kp + (long)(wave * 16 + lrow) * NSEQ;
  const _Float16* brow0 = vp + (long)(0 * 16 + lrow) * NSEQ;
  const _Float16* brow1 = vp + (long)(1 * 16 + lrow) * NSEQ;
  const _Float16* brow2 = vp + (long)(2 * 16 + lrow) * NSEQ;
  const _Float16* brow3 = vp + (long)(3 * 16 + lrow) * NSEQ;

  auto ldA = [&](int n0) {
    return frag16(*(const h8*)(arow + n0 + ah), *(const h8*)(arow + n0 + ah + 16));
  };
  auto ldB = [&](const _Float16* br, int n0) {
    return frag16(*(const h8*)(br + n0 + ah), *(const h8*)(br + n0 + ah + 16));
  };

  v8f acc[4] = {};

  // ping-pong buffers: X computes while Y loads, then swap (no reg moves)
  v16h afX = ldA(0);
  v16h bfX[4] = { ldB(brow0, 0), ldB(brow1, 0), ldB(brow2, 0), ldB(brow3, 0) };
  v16h afY, bfY[4];

  // NSEQ/32 = 128 iterations -> 64 double-iterations
  for (int n0 = 0; n0 < NSEQ; n0 += 64) {
    // issue loads for n0+32 into Y while computing on X
    afY = ldA(n0 + 32);
    bfY[0] = ldB(brow0, n0 + 32);
    bfY[1] = ldB(brow1, n0 + 32);
    bfY[2] = ldB(brow2, n0 + 32);
    bfY[3] = ldB(brow3, n0 + 32);
#pragma unroll
    for (int t = 0; t < 4; ++t)
      acc[t] = wmma16(afX, bfX[t], acc[t]);

    // issue loads for n0+64 into X (skip on last trip) while computing on Y
    if (n0 + 64 < NSEQ) {
      afX = ldA(n0 + 64);
      bfX[0] = ldB(brow0, n0 + 64);
      bfX[1] = ldB(brow1, n0 + 64);
      bfX[2] = ldB(brow2, n0 + 64);
      bfX[3] = ldB(brow3, n0 + 64);
    }
#pragma unroll
    for (int t = 0; t < 4; ++t)
      acc[t] = wmma16(afY, bfY[t], acc[t]);
  }

  _Float16* cp = ctxT + ((long)b * NH + h) * (DH * DH);
#pragma unroll
  for (int t = 0; t < 4; ++t)
#pragma unroll
    for (int r = 0; r < 8; ++r) {
      int d = wave * 16 + r + ah;
      int e = t * 16 + lrow;
      cp[(long)e * DH + d] = (_Float16)acc[t][r];   // transposed store
    }
}

// ---------------------------------------------------------------------------
// out[e,n] = sum_d ctxT[e,d] * q[d,n]  (M=64, N=4096 tiled by 256, K=64).
// ctxT fragments straight from global; q tile staged transposed in LDS.
// Writes f16 into the (dead) v region. 256 threads = 8 waves, wave w owns
// n-range w*32 (2 subtiles) x full M (4 subtiles).
// ---------------------------------------------------------------------------
__global__ __launch_bounds__(256)
void attnout_kernel(const _Float16* __restrict__ qkv,
                    const _Float16* __restrict__ ctxT) {
  const int nb = blockIdx.x, h = blockIdx.y, b = blockIdx.z;
  const _Float16* qp = qkv + ((long)b * QKV_M + h * DH) * NSEQ;       // q rows
  const _Float16* cp = ctxT + ((long)b * NH + h) * (DH * DH);
  _Float16* op = (_Float16*)qkv + ((long)b * QKV_M + 2 * HID + h * DH) * NSEQ;

  __shared__ __align__(16) _Float16 Qs[256][72];   // [n][d], 144B row stride
  const int tid = threadIdx.x;
  const int wave = tid >> 5, lane = tid & 31;
  const int lrow = lane & 15;
  const int ah = (lane < 16) ? 0 : 8;
  const int bh = (lane < 16) ? 0 : 16;
  const long n0 = (long)nb * 256;

#pragma unroll
  for (int i = 0; i < 8; ++i) {
    int f = tid + i * 256;          // 0..2047 chunks of 8 halves
    int d = f >> 5, n8 = (f & 31) << 3;
    h8 v = *(const h8*)(qp + (long)d * NSEQ + n0 + n8);
#pragma unroll
    for (int j = 0; j < 8; ++j) Qs[n8 + j][d] = v[j];
  }
  __syncthreads();

  v8f acc[4][2] = {};
#pragma unroll
  for (int kk = 0; kk < DH; kk += 32) {
    v16h af[4], bf[2];
#pragma unroll
    for (int s = 0; s < 4; ++s) {
      const _Float16* ar = cp + (long)(s * 16 + lrow) * DH + kk + ah;
      af[s] = frag16(*(const h8*)ar, *(const h8*)(ar + 16));
    }
#pragma unroll
    for (int t = 0; t < 2; ++t) {
      int n = wave * 32 + t * 16 + lrow;
      bf[t] = frag16(*(const h8*)&Qs[n][kk + ah], *(const h8*)&Qs[n][kk + ah + 16]);
    }
#pragma unroll
    for (int s = 0; s < 4; ++s)
#pragma unroll
      for (int t = 0; t < 2; ++t)
        acc[s][t] = wmma16(af[s], bf[t], acc[s][t]);
  }

#pragma unroll
  for (int s = 0; s < 4; ++s)
#pragma unroll
    for (int t = 0; t < 2; ++t)
#pragma unroll
      for (int r = 0; r < 8; ++r) {
        int e = s * 16 + r + ah;
        long n = n0 + wave * 32 + t * 16 + lrow;
        op[(long)e * NSEQ + n] = (_Float16)acc[s][t][r];
      }
}

// ---------------------------------------------------------------------------
extern "C" void kernel_launch(void* const* d_in, const int* in_sizes, int n_in,
                              void* d_out, int out_size, void* d_ws, size_t ws_size,
                              hipStream_t stream) {
  (void)in_sizes; (void)n_in; (void)out_size; (void)ws_size;
  const float* x     = (const float*)d_in[0];   // [16,256,64,64]
  const float* w_qkv = (const float*)d_in[1];   // [1536,256]
  const float* w_out = (const float*)d_in[2];   // [256,512]
  const float* b_out = (const float*)d_in[3];   // [256]
  float* out = (float*)d_out;                    // [16,256,64,64]

  _Float16* qkv  = (_Float16*)d_ws;                             // 192 MiB
  _Float16* ctxT = qkv + (size_t)NB * QKV_M * NSEQ;             // +1 MiB

  dim3 blk(256);
  // 1) qkv = w_qkv @ x  (per batch), f16 out, q rows scaled
  gemm_wmma_kernel<false, true><<<dim3(NSEQ / 128, QKV_M / 128, NB), blk, 0, stream>>>(
      w_qkv, x, qkv, nullptr, QKV_M, NSEQ, NCH,
      (long)NCH * NSEQ, (long)QKV_M * NSEQ);
  // 2) softmax over n on k region, in place
  softmax_rows_kernel<<<dim3(NB * HID), blk, 0, stream>>>(qkv);
  // 3) context[d,e] per (b,h), stored transposed
  context_kernel<<<dim3(NH, NB), dim3(128), 0, stream>>>(qkv, ctxT);
  // 4) out = ctxT @ q  -> overwrites v region
  attnout_kernel<<<dim3(NSEQ / 256, NH, NB), blk, 0, stream>>>(qkv, ctxT);
  // 5) final = w_out @ out + b_out  (per batch), fp32 out
  gemm_wmma_kernel<true, false><<<dim3(NSEQ / 128, NCH / 128, NB), blk, 0, stream>>>(
      w_out, qkv + (size_t)2 * HID * NSEQ, out, b_out, NCH, NSEQ, HID,
      (long)QKV_M * NSEQ, (long)NCH * NSEQ);
}
